// MultiHeadedAttention_49615462203796
// MI455X (gfx1250) — matrix operand order repaired and checked by reading, not compiled, not run on hardware
//
#include <hip/hip_runtime.h>
#include <hip/hip_bf16.h>

// ---------------------------------------------------------------------------
// MI455X (gfx1250) multi-head attention with 1.5-entmax, wave32 + WMMA bf16
// + TDM tensor_load_to_lds for the Q tile.
// B=2, S=2048, D=512, H=8, DK=64.
// ---------------------------------------------------------------------------

#define BATCH 2
#define SEQ   2048
#define DMODEL 512
#define NHEAD 8
#define DKH   64

typedef __attribute__((ext_vector_type(16))) __bf16 v16bf;
typedef __attribute__((ext_vector_type(8)))  __bf16 v8bf;
typedef __attribute__((ext_vector_type(8)))  float  v8f;
typedef __attribute__((ext_vector_type(4)))  float  v4f;
typedef __attribute__((ext_vector_type(4)))  unsigned int su32x4;
typedef __attribute__((ext_vector_type(8)))  unsigned int su32x8;

// D = A(16x32 bf16) x B(32x16 bf16) + C(16x16 f32), wave32.
__device__ __forceinline__ v8f wmma_bf16(v16bf a, v16bf b, v8f c) {
  return __builtin_amdgcn_wmma_f32_16x16x32_bf16(
      /*neg_a=*/false, a, /*neg_b=*/false, b,
      /*c_mod=*/(short)0, c, /*reuse_a=*/false, /*reuse_b=*/false);
}

// A-matrix 16x32 bf16 fragment: lane = M row.
// lane<16 holds K {0..15}; lane>=16 holds K {16..31}.
__device__ __forceinline__ v16bf load_a_bf16(const __bf16* rowk, int lane) {
  const int o0 = (lane < 16) ? 0 : 16;
  v8bf p0 = *(const v8bf*)(rowk + o0);
  v8bf p1 = *(const v8bf*)(rowk + o0 + 8);
  v16bf r;
#pragma unroll
  for (int i = 0; i < 8; ++i) { r[i] = p0[i]; r[i + 8] = p1[i]; }
  return r;
}

// Same A fragment but source is f32 (convert on the fly to bf16).
__device__ __forceinline__ v16bf load_a_f32(const float* rowk, int lane) {
  const int o0 = (lane < 16) ? 0 : 16;
  v16bf r;
#pragma unroll
  for (int i = 0; i < 8; ++i) r[i]     = (__bf16)rowk[o0 + i];
#pragma unroll
  for (int i = 0; i < 8; ++i) r[i + 8] = (__bf16)rowk[o0 + 8 + i];
  return r;
}

// ---------------------------------------------------------------------------
// Weight transpose + f32->bf16: Wt[n][k] = W[k][n], 512x512.
// ---------------------------------------------------------------------------
__global__ __launch_bounds__(256) void cvt_wT_kernel(const float* __restrict__ W,
                                                     __bf16* __restrict__ Wt) {
  int idx = blockIdx.x * 256 + threadIdx.x;      // over output [n][k]
  int n = idx >> 9, k = idx & 511;
  Wt[idx] = (__bf16)W[k * DMODEL + n];
}

// ---------------------------------------------------------------------------
// Projection GEMM: Out = (X[4096x512] @ W + bias) -> bf16, head-split layout.
// MODE 0: Q, scaled by 1/(2*sqrt(DK)) = 0.0625, layout [B,H,S,DK]
// MODE 1: K, layout [B,H,S,DK]
// MODE 2: V, layout [B,H,DK,S]  (transposed for contiguous PV B-fragments)
// ---------------------------------------------------------------------------
template <int MODE>
__global__ __launch_bounds__(256) void proj_kernel(const float* __restrict__ X,
                                                   const __bf16* __restrict__ Wt,
                                                   const float* __restrict__ bias,
                                                   __bf16* __restrict__ Out) {
  const int lane  = threadIdx.x & 31;
  const int wave  = threadIdx.x >> 5;
  const int mBase = blockIdx.x * 16;
  const int n0    = wave * 64;
  const int bsel  = (lane < 16) ? 0 : 16;
  const int mOff  = (lane < 16) ? 0 : 8;

  const float* xrow = X + (mBase + (lane & 15)) * DMODEL;
  v8f c[4] = {};
#pragma unroll 1
  for (int kk = 0; kk < DMODEL; kk += 32) {
    v16bf a = load_a_f32(xrow + kk, lane);
#pragma unroll
    for (int t = 0; t < 4; ++t) {
      const __bf16* bp = Wt + (n0 + t * 16 + (lane & 15)) * DMODEL + kk + bsel;
      v16bf b = *(const v16bf*)bp;           // contiguous 32B, B-layout
      c[t] = wmma_bf16(a, b, c[t]);
    }
  }
  const float scale = (MODE == 0) ? 0.0625f : 1.0f;
#pragma unroll
  for (int t = 0; t < 4; ++t) {
    const int nGlob = n0 + t * 16 + (lane & 15);
    const float bv = bias[nGlob];
    const int h = nGlob >> 6, dk = nGlob & 63;
#pragma unroll
    for (int r = 0; r < 8; ++r) {
      const int mGlob = mBase + mOff + r;
      const int bi = mGlob >> 11, si = mGlob & 2047;
      const float v = (c[t][r] + bv) * scale;
      if (MODE == 2)
        Out[((bi * NHEAD + h) * DKH + dk) * SEQ + si] = (__bf16)v;
      else
        Out[((bi * NHEAD + h) * SEQ + si) * DKH + dk] = (__bf16)v;
    }
  }
}

// ---------------------------------------------------------------------------
// Attention: per (b,h, 16-query tile).
// Phase 0: TDM tensor_load_to_lds stages the 16x64 bf16 Q tile into LDS.
// Phase 1: scores y = (Q Kᵀ)/(2 sqrt(DK)) into LDS (16 x 2048 f32) via WMMA.
// Phase 2: 1.5-entmax per row; row chunk held in 64 VGPRs/lane, tau by
//          22-step bisection on sum(clip(y-tau,0)^2)=1 (monotone in tau).
// Phase 3: ctx = P @ V via WMMA (P bf16 from LDS, V from transposed global).
// ---------------------------------------------------------------------------
#define YSTRIDE 2056
#define PSTRIDE 2064
#define QLDS_OFF (16 * YSTRIDE * 4 + 16 * PSTRIDE * 2 + 8 * 16 * 16 * 4)
#define ATTN_LDS_BYTES (QLDS_OFF + 16 * DKH * 2)

__global__ __launch_bounds__(256) void attn_kernel(const __bf16* __restrict__ Qb,
                                                   const __bf16* __restrict__ Kb,
                                                   const __bf16* __restrict__ Vt,
                                                   __bf16* __restrict__ Ctx) {
  extern __shared__ char smem[];
  float  (*s_y)[YSTRIDE] = (float (*)[YSTRIDE])smem;
  __bf16 (*s_p)[PSTRIDE] = (__bf16 (*)[PSTRIDE])(smem + 16 * YSTRIDE * 4);
  float  (*s_part)[16][16] =
      (float (*)[16][16])(smem + 16 * YSTRIDE * 4 + 16 * PSTRIDE * 2);
  __bf16* q_lds = (__bf16*)(smem + QLDS_OFF);

  const int lane = threadIdx.x & 31;
  const int wave = threadIdx.x >> 5;
  const int tile = blockIdx.x;
  const int qt = tile & 127;                 // S/16 tiles
  const int bh = tile >> 7;                  // 0..B*H-1
  const int q0 = qt * 16;
  const int bsel = (lane < 16) ? 0 : 16;
  const int mOff = (lane < 16) ? 0 : 8;

  const __bf16* Qhead = Qb + (size_t)bh * SEQ * DKH;
  const __bf16* Khead = Kb + (size_t)bh * SEQ * DKH;
  const __bf16* Vhead = Vt + (size_t)bh * DKH * SEQ;

  // ---- Phase 0: one TDM op stages the Q tile (16 rows x 64 bf16). ----
  if (wave == 0) {
    const __bf16* gq = Qhead + (size_t)q0 * DKH;
    const unsigned long long ga = (unsigned long long)(uintptr_t)gq;
    const unsigned int lds_off = (unsigned int)(uintptr_t)q_lds; // LDS byte off
    su32x4 g0;
    g0[0] = 1u;                                   // count=1, user descriptor
    g0[1] = lds_off;                              // lds_addr
    g0[2] = (unsigned int)ga;                     // global_addr[31:0]
    g0[3] = (unsigned int)(ga >> 32) | (2u << 30);// global_addr[56:32]|type=2
    su32x8 g1;
    g1[0] = (1u << 16);          // wg_mask=0, data_size=1 (2 bytes)
    g1[1] = (64u << 16);         // tensor_dim0 = 64 (low 16 bits)
    g1[2] = ((unsigned)SEQ << 16); // tensor_dim0 hi=0, tensor_dim1 lo = 2048
    g1[3] = (64u << 16);         // tensor_dim1 hi=0, tile_dim0 = 64
    g1[4] = 16u;                 // tile_dim1 = 16, tile_dim2 = 0
    g1[5] = 64u;                 // tensor_dim0_stride = 64
    g1[6] = 0u;                  // stride0 hi, tensor_dim1_stride lo
    g1[7] = 0u;                  // tensor_dim1_stride hi
    asm volatile("tensor_load_to_lds %0, %1" :: "s"(g0), "s"(g1) : "memory");
    __builtin_amdgcn_s_wait_tensorcnt(0);
  }
  __syncthreads();

  // Q A-fragments from LDS, both 32-wide K windows of DK=64.
  const __bf16* qrow = q_lds + (lane & 15) * DKH;
  const v16bf aq0 = load_a_bf16(qrow, lane);
  const v16bf aq1 = load_a_bf16(qrow + 32, lane);

  // ---- Phase 1: scores strip. Wave w covers keys [w*256, w*256+256). ----
#pragma unroll 1
  for (int nt = 0; nt < 16; ++nt) {
    const int col0 = wave * 256 + nt * 16;
    const __bf16* kp = Khead + (col0 + (lane & 15)) * DKH + bsel;
    __builtin_prefetch((const void*)(kp + 16 * DKH), 0, 0);  // next tile
    v16bf b0 = *(const v16bf*)kp;
    v16bf b1 = *(const v16bf*)(kp + 32);
    v8f c = {};
    c = wmma_bf16(aq0, b0, c);
    c = wmma_bf16(aq1, b1, c);
    const int col = col0 + (lane & 15);
#pragma unroll
    for (int r = 0; r < 8; ++r) s_y[mOff + r][col] = c[r];
  }
  __syncthreads();

  // ---- Phase 2: entmax-1.5, 2 rows per wave, row chunk in registers. ----
#pragma unroll 1
  for (int rr = 0; rr < 2; ++rr) {
    const int row = wave * 2 + rr;
    const float* yrow = s_y[row] + lane * 64;   // 64 contiguous f32 per lane
    v4f yv[16];
#pragma unroll
    for (int j = 0; j < 16; ++j) yv[j] = *(const v4f*)(yrow + j * 4);

    float mx = -3.0e38f;
#pragma unroll
    for (int j = 0; j < 16; ++j)
      mx = fmaxf(mx, fmaxf(fmaxf(yv[j][0], yv[j][1]), fmaxf(yv[j][2], yv[j][3])));
#pragma unroll
    for (int off = 16; off > 0; off >>= 1) mx = fmaxf(mx, __shfl_xor(mx, off, 32));

    float lo = mx - 1.0f, hi = mx;              // tau bracket (support >= 1)
#pragma unroll 1
    for (int it = 0; it < 22; ++it) {           // monotone bisection for tau*
      const float mid = 0.5f * (lo + hi);
      float s = 0.0f;
#pragma unroll
      for (int j = 0; j < 16; ++j)
#pragma unroll
        for (int e = 0; e < 4; ++e) {
          const float d = fmaxf(yv[j][e] - mid, 0.0f);
          s = fmaf(d, d, s);
        }
#pragma unroll
      for (int off = 16; off > 0; off >>= 1) s += __shfl_xor(s, off, 32);
      if (s >= 1.0f) lo = mid; else hi = mid;
    }
    const float tau = 0.5f * (lo + hi);

    __bf16* prow = s_p[row] + lane * 64;
#pragma unroll
    for (int j = 0; j < 8; ++j) {               // packed b128 stores of p=bf16
      v8bf pb;
#pragma unroll
      for (int e = 0; e < 8; ++e) {
        const int q = j * 8 + e;
        const float d = fmaxf(yv[q >> 2][q & 3] - tau, 0.0f);
        pb[e] = (__bf16)(d * d);
      }
      *(v8bf*)(prow + j * 8) = pb;
    }
  }
  __syncthreads();

  // ---- Phase 3: ctx = P @ V.  wave = (key-half)*4 + feature-tile ----
  const int ft = wave & 3;                   // feature tile (16 of DK=64)
  const int kh = wave >> 2;                  // key half (1024 each)
  const int f0 = ft * 16;
  const int m = lane & 15;
  const __bf16* vrow = Vhead + (f0 + (lane & 15)) * SEQ;  // Vt[feat][key]
  v8f c = {};
#pragma unroll 1
  for (int ks = 0; ks < 32; ++ks) {
    const int kb = kh * 1024 + ks * 32;
    __builtin_prefetch((const void*)(vrow + kb + 256), 0, 0);
    v16bf a = load_a_bf16(&s_p[m][kb], lane);
    v16bf b = *(const v16bf*)(vrow + kb + bsel);
    c = wmma_bf16(a, b, c);
  }
#pragma unroll
  for (int r = 0; r < 8; ++r) s_part[wave][mOff + r][lane & 15] = c[r];
  __syncthreads();

  // Waves 0..3 reduce the two key-halves and emit ctx [B,S,H*DK].
  if (wave < 4) {
    const int bi = bh >> 3, h = bh & 7;
#pragma unroll
    for (int e = 0; e < 8; ++e) {
      const int idx = e * 32 + lane;
      const int i = idx >> 4, j = idx & 15;
      const float v = s_part[wave][i][j] + s_part[wave + 4][i][j];
      Ctx[((size_t)(bi * SEQ + q0 + i)) * DMODEL + h * DKH + wave * 16 + j] =
          (__bf16)v;
    }
  }
}

// ---------------------------------------------------------------------------
// Output projection: out_f32 = Ctx(bf16) @ Wo + bo.
// ---------------------------------------------------------------------------
__global__ __launch_bounds__(256) void outproj_kernel(const __bf16* __restrict__ Ctx,
                                                      const __bf16* __restrict__ Wt,
                                                      const float* __restrict__ bias,
                                                      float* __restrict__ Out) {
  const int lane  = threadIdx.x & 31;
  const int wave  = threadIdx.x >> 5;
  const int mBase = blockIdx.x * 16;
  const int n0    = wave * 64;
  const int bsel  = (lane < 16) ? 0 : 16;
  const int mOff  = (lane < 16) ? 0 : 8;

  const __bf16* crow = Ctx + (size_t)(mBase + (lane & 15)) * DMODEL;
  v8f c[4] = {};
#pragma unroll 1
  for (int kk = 0; kk < DMODEL; kk += 32) {
    v16bf a = load_a_bf16(crow + kk, lane);
#pragma unroll
    for (int t = 0; t < 4; ++t) {
      const __bf16* bp = Wt + (n0 + t * 16 + (lane & 15)) * DMODEL + kk + bsel;
      v16bf b = *(const v16bf*)bp;
      c[t] = wmma_bf16(a, b, c[t]);
    }
  }
#pragma unroll
  for (int t = 0; t < 4; ++t) {
    const int nGlob = n0 + t * 16 + (lane & 15);
    const float bv = bias[nGlob];
#pragma unroll
    for (int r = 0; r < 8; ++r) {
      const int mGlob = mBase + mOff + r;
      Out[(size_t)mGlob * DMODEL + nGlob] = c[t][r] + bv;
    }
  }
}

// ---------------------------------------------------------------------------
extern "C" void kernel_launch(void* const* d_in, const int* in_sizes, int n_in,
                              void* d_out, int out_size, void* d_ws, size_t ws_size,
                              hipStream_t stream) {
  const float* query = (const float*)d_in[0];
  const float* key   = (const float*)d_in[1];
  const float* value = (const float*)d_in[2];
  const float* Wq = (const float*)d_in[3];
  const float* bq = (const float*)d_in[4];
  const float* Wk = (const float*)d_in[5];
  const float* bk = (const float*)d_in[6];
  const float* Wv = (const float*)d_in[7];
  const float* bv = (const float*)d_in[8];
  const float* Wo = (const float*)d_in[9];
  const float* bo = (const float*)d_in[10];
  float* out = (float*)d_out;

  const size_t WELEM = (size_t)DMODEL * DMODEL;               // 262144
  const size_t HELEM = (size_t)BATCH * NHEAD * SEQ * DKH;     // 2097152
  __bf16* WqT = (__bf16*)d_ws;
  __bf16* WkT = WqT + WELEM;
  __bf16* WvT = WkT + WELEM;
  __bf16* WoT = WvT + WELEM;
  __bf16* Qb  = WoT + WELEM;
  __bf16* Kb  = Qb + HELEM;
  __bf16* Vt  = Kb + HELEM;
  __bf16* Ctx = Vt + HELEM;                                   // total ~18 MB

  const dim3 blk(256);
  cvt_wT_kernel<<<WELEM / 256, blk, 0, stream>>>(Wq, WqT);
  cvt_wT_kernel<<<WELEM / 256, blk, 0, stream>>>(Wk, WkT);
  cvt_wT_kernel<<<WELEM / 256, blk, 0, stream>>>(Wv, WvT);
  cvt_wT_kernel<<<WELEM / 256, blk, 0, stream>>>(Wo, WoT);

  proj_kernel<0><<<(BATCH * SEQ) / 16, blk, 0, stream>>>(query, WqT, bq, Qb);
  proj_kernel<1><<<(BATCH * SEQ) / 16, blk, 0, stream>>>(key,   WkT, bk, Kb);
  proj_kernel<2><<<(BATCH * SEQ) / 16, blk, 0, stream>>>(value, WvT, bv, Vt);

  attn_kernel<<<BATCH * NHEAD * (SEQ / 16), blk, ATTN_LDS_BYTES, stream>>>(
      Qb, Kb, Vt, Ctx);

  outproj_kernel<<<(BATCH * SEQ) / 16, blk, 0, stream>>>(Ctx, WoT, bo, out);
}
